// GTGModule_24541443129749
// MI455X (gfx1250) — matrix operand order repaired and checked by reading, not compiled, not run on hardware
//
#include <hip/hip_runtime.h>
#include <math.h>
#include <stdint.h>

#define BB 2048
#define NCLS 10
#define MAXIT 30
#define TOL 0.001f
#define BN_EPS 1e-5f

typedef __attribute__((ext_vector_type(16))) __bf16 bf16x16;
typedef __attribute__((ext_vector_type(8)))  float  f32x8;
typedef __attribute__((ext_vector_type(2)))  float  f32x2;

union BF16Tile { bf16x16 v; uint4 q[2]; };

__device__ __forceinline__ float warp_reduce_add(float s) {
#pragma unroll
  for (int off = 16; off; off >>= 1) s += __shfl_xor(s, off, 32);
  return s;
}

// ---------------- small utility kernels ----------------
__global__ void k_zero(float* p, int n) {
  int i = blockIdx.x * blockDim.x + threadIdx.x;
  if (i < n) p[i] = 0.0f;
}

__global__ void k_copy(float* dst, const float* src, int n) {
  int i = blockIdx.x * blockDim.x + threadIdx.x;
  if (i < n) dst[i] = src[i];
}

__global__ void k_scatter_lab(const int* idx, int n, float* labm) {
  int i = blockIdx.x * blockDim.x + threadIdx.x;
  if (i < n) labm[idx[i]] = 1.0f;
}

// X0: labelled rows = softmax(outs); unlabelled rows = 1/NC; padded to 16 cols
__global__ void k_x0(const float* outs, const float* labm, float* X0p) {
  int row = blockIdx.x * blockDim.x + threadIdx.x;
  if (row >= BB) return;
  float* x = X0p + row * 16;
  if (labm[row] > 0.5f) {
    const float* o = outs + row * NCLS;
    float m = o[0];
#pragma unroll
    for (int c = 1; c < NCLS; ++c) m = fmaxf(m, o[c]);
    float s = 0.f, e[NCLS];
#pragma unroll
    for (int c = 0; c < NCLS; ++c) { e[c] = __expf(o[c] - m); s += e[c]; }
    float inv = 1.0f / s;
#pragma unroll
    for (int c = 0; c < NCLS; ++c) x[c] = e[c] * inv;
  } else {
#pragma unroll
    for (int c = 0; c < NCLS; ++c) x[c] = 1.0f / NCLS;
  }
#pragma unroll
  for (int c = NCLS; c < 16; ++c) x[c] = 0.0f;
}

// ---------------- pooling + embed ----------------
// One wave per (b,c) pair: lane-strided (fully coalesced) loads + wave reduction.
__global__ void k_pool(const float* __restrict__ f, float* pooled, int C, int FF, float inv) {
  int wavesPerBlock = blockDim.x >> 5;
  int pair = blockIdx.x * wavesPerBlock + (threadIdx.x >> 5);
  int lane = threadIdx.x & 31;
  if (pair >= BB * C) return;
  const float* p = f + (size_t)pair * FF;
  float s = 0.f;
  for (int t = lane; t < FF; t += 32) s += p[t];
  s = warp_reduce_add(s);
  if (lane == 0) pooled[pair] = s * inv;
}

__global__ void k_emb(const float* __restrict__ pooled, const float* __restrict__ Wl,
                      const float* __restrict__ bl, float* emb, int C) {
  int idx = blockIdx.x * blockDim.x + threadIdx.x;
  if (idx >= BB * 128) return;
  int b = idx >> 7, j = idx & 127;
  const float* pr = pooled + (size_t)b * C;
  __builtin_prefetch(pr, 0, 0);
  float s = bl[j];
  for (int c = 0; c < C; ++c) s += pr[c] * Wl[c * 128 + j];
  emb[idx] = fmaxf(s, 0.0f);
}

// row-normalize then convert to bf16 (RNE)
__global__ void k_normalize_bf16(const float* __restrict__ E, uint16_t* Nbf, int D) {
  int row = blockIdx.x, t = threadIdx.x;
  __shared__ float red[128];
  float s = 0.f;
  for (int k = t; k < D; k += 128) { float v = E[(size_t)row * D + k]; s += v * v; }
  red[t] = s; __syncthreads();
  for (int o = 64; o; o >>= 1) { if (t < o) red[t] += red[t + o]; __syncthreads(); }
  float inv = 1.0f / fmaxf(sqrtf(red[0]), 1e-12f);
  for (int k = t; k < D; k += 128) {
    float v = E[(size_t)row * D + k] * inv;
    uint32_t u = __float_as_uint(v);
    uint32_t r = (u + 0x7FFFu + ((u >> 16) & 1u)) >> 16;
    Nbf[(size_t)row * D + k] = (uint16_t)r;
  }
}

// ---------------- A = N @ N^T via bf16 WMMA; fused clamp/diag/mean-sum ----------------
// Each wave computes one 16x32 output strip (two 16x16 tiles): the A-operand
// fragment is loaded once per K-step and reused across both WMMAs.
__global__ void k_build_a(const uint16_t* __restrict__ Nbf, int D, float* A, float* scal) {
  int mi = blockIdx.x;
  int nj0 = blockIdx.y * 2, nj1 = nj0 + 1;
  int lane = threadIdx.x;
  int rowA  = 16 * mi  + (lane & 15);
  int rowB0 = 16 * nj0 + (lane & 15);
  int rowB1 = rowB0 + 16;
  int aoff = (lane < 16) ? 0 : 16;   // A 16x32 bf16 lane layout: K {0..15}/{16..31}, contiguous
  int boff = (lane < 16) ? 0 : 16;   // B 32x16 bf16 lane layout: K {0..15}/{16..31}, contiguous
  f32x8 c0, c1;
#pragma unroll
  for (int v = 0; v < 8; ++v) { c0[v] = 0.0f; c1[v] = 0.0f; }
  for (int kb = 0; kb < D; kb += 32) {
    BF16Tile a, b0, b1;
    const uint16_t* pa = Nbf + (size_t)rowA * D + kb + aoff;
    a.q[0] = *(const uint4*)(pa);
    a.q[1] = *(const uint4*)(pa + 8);
    const uint16_t* pb0 = Nbf + (size_t)rowB0 * D + kb + boff;
    b0.q[0] = *(const uint4*)(pb0);
    b0.q[1] = *(const uint4*)(pb0 + 8);
    const uint16_t* pb1 = Nbf + (size_t)rowB1 * D + kb + boff;
    b1.q[0] = *(const uint4*)(pb1);
    b1.q[1] = *(const uint4*)(pb1 + 8);
    c0 = __builtin_amdgcn_wmma_f32_16x16x32_bf16(false, a.v, false, b0.v, (short)0, c0, false, false);
    c1 = __builtin_amdgcn_wmma_f32_16x16x32_bf16(false, a.v, false, b1.v, (short)0, c1, false, false);
  }
  int m0 = (lane < 16) ? 0 : 8;
  int col0 = 16 * nj0 + (lane & 15);
  int col1 = col0 + 16;
  float s = 0.f;
#pragma unroll
  for (int v = 0; v < 8; ++v) {
    int r = 16 * mi + v + m0;
    float v0 = (r == col0) ? 0.0f : fminf(fmaxf(c0[v], 0.0f), 1.0f);
    float v1 = (r == col1) ? 0.0f : fminf(fmaxf(c1[v], 0.0f), 1.0f);
    A[(size_t)r * BB + col0] = v0;
    A[(size_t)r * BB + col1] = v1;
    s += v0 + v1;
  }
  s = warp_reduce_add(s);
  if (lane == 0) atomicAdd(&scal[0], s);
}

__global__ void k_finalize_a(float* A, const float* __restrict__ labm, const float* scal) {
  int idx = blockIdx.x * blockDim.x + threadIdx.x;
  if (idx >= BB * BB) return;
  float thresh = scal[0] * (1.0f / ((float)BB * (float)BB));
  int i = idx / BB, j = idx - i * BB;
  float a = A[idx];
  float v = (a < thresh) ? 0.0f : a;
  bool bothlab = (labm[i] > 0.5f) && (labm[j] > 0.5f);
  A[idx] = bothlab ? v : 1.0f - v;   // A=1-A everywhere, labelled x labelled flipped back
}

// ---------------- GTG: AX = A @ Xp (f32 WMMA 16x16x4, X async-staged to LDS) -----------
__global__ void k_gtg_mm(const float* __restrict__ A, const float* __restrict__ Xp,
                         float* AXp, float* scal, int iter) {
  extern __shared__ float smem[];    // 2048 x 16 f32 = 128 KB
  if (blockIdx.x == 0 && threadIdx.x == 0 && iter > 0) {
    // consume err from previous iteration, update active flag, reset err
    float e = scal[1];
    if (!(sqrtf(e) > TOL)) ((int*)scal)[2] = 0;
    scal[1] = 0.0f;
  }
  // async-stage all of X (2048x16 f32) into LDS: 8192 x b128 chunks
  for (int c = threadIdx.x; c < (BB * 16 * 4) / 16; c += blockDim.x) {
    uint32_t lds_off = (uint32_t)(c * 16);
    uint64_t gaddr = (uint64_t)((const char*)Xp + (size_t)c * 16);
    asm volatile("global_load_async_to_lds_b128 %0, %1, off"
                 :: "v"(lds_off), "v"(gaddr) : "memory");
  }
  asm volatile("s_wait_asynccnt 0" ::: "memory");
  __syncthreads();

  int wave = threadIdx.x >> 5, lane = threadIdx.x & 31;
  int mi = blockIdx.x;
  int row = 16 * mi + (lane & 15);
  int klo = (lane < 16) ? 0 : 2;     // f32 16x4 A / 4x16 B lane layout
  int col = lane & 15;
  const float* Arow = A + (size_t)row * BB;
  f32x8 c;
#pragma unroll
  for (int v = 0; v < 8; ++v) c[v] = 0.0f;
  int k0 = wave * (BB / 4), k1 = k0 + (BB / 4);
  for (int kb = k0; kb < k1; kb += 4) {
    f32x2 a = *(const f32x2*)(Arow + kb + klo);
    f32x2 b;
    b.x = smem[(kb + klo + 0) * 16 + col];
    b.y = smem[(kb + klo + 1) * 16 + col];
    c = __builtin_amdgcn_wmma_f32_16x16x4_f32(false, a, false, b, (short)0, c, false, false);
  }
  __syncthreads();                    // done reading X; reuse LDS for split-K reduction
  if (wave > 0) {
#pragma unroll
    for (int v = 0; v < 8; ++v) smem[((wave - 1) * 32 + lane) * 8 + v] = c[v];
  }
  __syncthreads();
  if (wave == 0) {
#pragma unroll
    for (int v = 0; v < 8; ++v)
      c[v] += smem[(lane) * 8 + v] + smem[(32 + lane) * 8 + v] + smem[(64 + lane) * 8 + v];
    int m0 = (lane < 16) ? 0 : 8;
#pragma unroll
    for (int v = 0; v < 8; ++v)
      AXp[(size_t)(16 * mi + v + m0) * 16 + col] = c[v];
  }
}

__global__ void k_gtg_reset(float* entAcc, float* scal) {
  int i = blockIdx.x * blockDim.x + threadIdx.x;
  if (i < BB) entAcc[i] = 0.0f;
  if (i == 0) { scal[1] = 0.0f; ((int*)scal)[2] = 1; }
}

__global__ void k_gtg_upd(float* Xp, const float* __restrict__ AXp, float* entAcc, float* scal) {
  int row = blockIdx.x * blockDim.x + threadIdx.x;
  if (row >= BB) return;
  int active = ((const int*)scal)[2];
  const float* ax = AXp + row * 16;
  float* x = Xp + row * 16;
  float M[NCLS]; float ssum = 0.f;
#pragma unroll
  for (int c = 0; c < NCLS; ++c) { M[c] = x[c] * ax[c]; ssum += M[c]; }
  float inv = 1.0f / fmaxf(ssum, 1e-30f);
  float errc = 0.f, ent = 0.f, xn[NCLS];
#pragma unroll
  for (int c = 0; c < NCLS; ++c) {
    xn[c] = M[c] * inv;
    float d = xn[c] - x[c];
    errc += d * d;
    float xc = fminf(fmaxf(xn[c], 1e-8f), 1.0f);
    ent -= xc * __logf(xc);
  }
  if (active) {
#pragma unroll
    for (int c = 0; c < NCLS; ++c) x[c] = xn[c];
    entAcc[row] += ent;
  }
  atomicAdd(&scal[1], errc);
}

// ---------------- MLP chain ----------------
__global__ void k_zinit(float* zbuf, const float* __restrict__ emb, const float* __restrict__ entAcc) {
  int idx = blockIdx.x * blockDim.x + threadIdx.x;
  if (idx >= BB * 129) return;
  int row = idx / 129, k = idx - row * 129;
  zbuf[idx] = (k < 128) ? emb[row * 128 + k] : entAcc[row] * (1.0f / MAXIT);
}

__global__ void k_gemm_bias(const float* __restrict__ Z, const float* __restrict__ Wm,
                            const float* __restrict__ bv, float* H, int din, int dout) {
  int idx = blockIdx.x * blockDim.x + threadIdx.x;
  if (idx >= BB * dout) return;
  int row = idx / dout, j = idx - row * dout;
  const float* zr = Z + (size_t)row * din;
  float s = bv[j];
  for (int k = 0; k < din; ++k) s += zr[k] * Wm[k * dout + j];
  H[idx] = s;
}

__global__ void k_bnstats(const float* __restrict__ H, float* mu, float* var, int dout) {
  int j = blockIdx.x, t = threadIdx.x;
  __shared__ float rs[256], rs2[256];
  float s = 0.f, s2 = 0.f;
  for (int r = t; r < BB; r += 256) {
    float v = H[(size_t)r * dout + j];
    s += v; s2 += v * v;
  }
  rs[t] = s; rs2[t] = s2; __syncthreads();
  for (int o = 128; o; o >>= 1) { if (t < o) { rs[t] += rs[t + o]; rs2[t] += rs2[t + o]; } __syncthreads(); }
  if (t == 0) {
    float m = rs[0] * (1.0f / BB);
    mu[j] = m;
    var[j] = rs2[0] * (1.0f / BB) - m * m;
  }
}

__global__ void k_bnapply(const float* __restrict__ H, const float* __restrict__ mu,
                          const float* __restrict__ var, const float* __restrict__ g,
                          const float* __restrict__ be, float* outp,
                          int dout, int outPitch, int colOff) {
  int idx = blockIdx.x * blockDim.x + threadIdx.x;
  if (idx >= BB * dout) return;
  int row = idx / dout, j = idx - row * dout;
  float v = (H[idx] - mu[j]) * rsqrtf(var[j] + BN_EPS) * g[j] + be[j];
  outp[(size_t)row * outPitch + colOff + j] = fmaxf(v, 0.0f);
}

// ---------------- outputs ----------------
__global__ void k_final(const float* __restrict__ cat, const float* __restrict__ fw,
                        const float* __restrict__ fb, float* out) {
  int row = blockIdx.x * blockDim.x + threadIdx.x;
  if (row >= BB) return;
  const float* cr = cat + (size_t)row * 128;
  float s = fb[0];
#pragma unroll 4
  for (int k = 0; k < 128; ++k) s += cr[k] * fw[k];
  out[row] = s;
}

__global__ void k_ytrue(const float* __restrict__ entAcc, float* out) {
  int row = blockIdx.x * blockDim.x + threadIdx.x;
  if (row < BB) out[BB + row] = entAcc[row] * (1.0f / MAXIT);
}

__global__ void k_mask(const float* __restrict__ labm, float* out) {
  int row = blockIdx.x * blockDim.x + threadIdx.x;
  if (row < BB) out[2 * BB + row] = labm[row];
}

// ---------------- host orchestration ----------------
static void build_A(const uint16_t* Nbf, int D, float* A, const float* labm, float* scal,
                    hipStream_t s) {
  k_zero<<<1, 32, 0, s>>>(scal, 1);
  k_build_a<<<dim3(128, 64), 32, 0, s>>>(Nbf, D, A, scal);
  k_finalize_a<<<(BB * BB) / 256, 256, 0, s>>>(A, labm, scal);
}

static void run_gtg(const float* A, float* Xp, const float* X0p, float* AXp,
                    float* entAcc, float* scal, hipStream_t s) {
  k_copy<<<(BB * 16 + 255) / 256, 256, 0, s>>>(Xp, X0p, BB * 16);
  k_gtg_reset<<<(BB + 255) / 256, 256, 0, s>>>(entAcc, scal);
  for (int it = 0; it < MAXIT; ++it) {
    k_gtg_mm<<<128, 128, BB * 16 * sizeof(float), s>>>(A, Xp, AXp, scal, it);
    k_gtg_upd<<<(BB + 255) / 256, 256, 0, s>>>(Xp, AXp, entAcc, scal);
  }
}

extern "C" void kernel_launch(void* const* d_in, const int* in_sizes, int n_in,
                              void* d_out, int out_size, void* d_ws, size_t ws_size,
                              hipStream_t stream) {
  const int CH[4] = {64, 128, 256, 512};
  const int FS[4] = {32, 16, 8, 4};
  const int MLP_DIMS[5][2] = {{129, 129}, {129, 64}, {64, 64}, {64, 32}, {32, 32}};

  const float* feats[4] = {(const float*)d_in[0], (const float*)d_in[1],
                           (const float*)d_in[2], (const float*)d_in[3]};
  const float* embedds = (const float*)d_in[4];
  const float* outs    = (const float*)d_in[5];
  const int*   labIdx  = (const int*)d_in[7];
  const int    nLab    = in_sizes[7];

  float* W = (float*)d_ws;
  float* A      = W;                         // 2048*2048
  float* AXp    = A + (size_t)BB * BB;       // 2048*16
  float* Xp     = AXp + BB * 16;
  float* X0p    = Xp + BB * 16;
  float* embBuf = X0p + BB * 16;             // 4 * 2048*128
  uint16_t* Nbf = (uint16_t*)(embBuf + 4 * (size_t)BB * 128);   // 2048*512 bf16
  float* pooled = (float*)(Nbf) + (BB * 512) / 2;               // 2048*512 f32
  float* entAcc = pooled + (size_t)BB * 512; // 2048
  float* zbuf   = entAcc + BB;               // 2048*129
  float* hbuf   = zbuf + (size_t)BB * 129;   // 2048*129
  float* mu     = hbuf + (size_t)BB * 129;   // 160
  float* var    = mu + 160;                  // 160
  float* cat    = var + 160;                 // 2048*128
  float* labm   = cat + (size_t)BB * 128;    // 2048
  float* scal   = labm + BB;                 // 64

  float* out = (float*)d_out;

  // labelled mask + X0
  k_zero<<<(BB + 255) / 256, 256, 0, stream>>>(labm, BB);
  k_scatter_lab<<<(nLab + 255) / 256, 256, 0, stream>>>(labIdx, nLab, labm);
  k_x0<<<(BB + 255) / 256, 256, 0, stream>>>(outs, labm, X0p);

  // branches
  for (int br = 0; br < 4; ++br) {
    int C = CH[br], F = FS[br], FF = F * F;
    const float* Wl = (const float*)d_in[9 + 2 * br];
    const float* bl = (const float*)d_in[10 + 2 * br];
    float* emb = embBuf + (size_t)br * BB * 128;

    k_pool<<<(BB * C) / 8, 256, 0, stream>>>(feats[br], pooled, C, FF, 1.0f / FF);
    k_emb<<<(BB * 128 + 255) / 256, 256, 0, stream>>>(pooled, Wl, bl, emb, C);
    k_normalize_bf16<<<BB, 128, 0, stream>>>(emb, Nbf, 128);
    build_A(Nbf, 128, A, labm, scal, stream);
    run_gtg(A, Xp, X0p, AXp, entAcc, scal, stream);

    // MLP with batchnorm
    k_zinit<<<(BB * 129 + 255) / 256, 256, 0, stream>>>(zbuf, emb, entAcc);
    int din = 129;
    for (int l = 0; l < 5; ++l) {
      int dout = MLP_DIMS[l][1];
      const float* Wm = (const float*)d_in[17 + br * 20 + l * 4 + 0];
      const float* bv = (const float*)d_in[17 + br * 20 + l * 4 + 1];
      const float* g  = (const float*)d_in[17 + br * 20 + l * 4 + 2];
      const float* be = (const float*)d_in[17 + br * 20 + l * 4 + 3];
      k_gemm_bias<<<(BB * dout + 255) / 256, 256, 0, stream>>>(zbuf, Wm, bv, hbuf, din, dout);
      k_bnstats<<<dout, 256, 0, stream>>>(hbuf, mu, var, dout);
      int outPitch = (l == 4) ? 128 : dout;
      int colOff   = (l == 4) ? br * 32 : 0;
      float* outp  = (l == 4) ? cat : zbuf;
      k_bnapply<<<(BB * dout + 255) / 256, 256, 0, stream>>>(hbuf, mu, var, g, be, outp,
                                                             dout, outPitch, colOff);
      din = dout;
    }
  }

  // y_pred
  k_final<<<(BB + 255) / 256, 256, 0, stream>>>(cat, (const float*)d_in[97],
                                                (const float*)d_in[98], out);

  // y_true: GTG on embedds (D=512)
  k_normalize_bf16<<<BB, 128, 0, stream>>>(embedds, Nbf, 512);
  build_A(Nbf, 512, A, labm, scal, stream);
  run_gtg(A, Xp, X0p, AXp, entAcc, scal, stream);
  k_ytrue<<<(BB + 255) / 256, 256, 0, stream>>>(entAcc, out);

  // mask
  k_mask<<<(BB + 255) / 256, 256, 0, stream>>>(labm, out);
}
